// SelfAttn_2293512536165
// MI455X (gfx1250) — compile-verified, matrix operand order checked by reading, not run on hardware
//
#include <hip/hip_runtime.h>

// ---------------------------------------------------------------------------
// CDNA5 (gfx1250) fused T5 self-attention.
// B=4, S=2048, D=512, H=8, HD=64.
// Matrix math: v_wmma_f32_16x16x32_bf16.  LDS staging: global_load_async_to_lds
// (ASYNCcnt path) instead of global->VGPR->ds_store.
// ---------------------------------------------------------------------------

typedef __attribute__((ext_vector_type(16))) __bf16 v16bf;
typedef __attribute__((ext_vector_type(8)))  float  v8f;

#define B_  4
#define S_  2048
#define D_  512
#define H_  8
#define HD_ 64
#define M_  (B_ * S_)   // 8192 rows of x

// ---- bf16 helpers (bit-level, round-to-nearest-even) ----------------------
__device__ __forceinline__ unsigned short f2bf_us(float f) {
  unsigned int u = __float_as_uint(f);
  unsigned int r = u + 0x7FFFu + ((u >> 16) & 1u);
  return (unsigned short)(r >> 16);
}
__device__ __forceinline__ __bf16 us2bf(unsigned short u) {
  union { unsigned short s; __bf16 b; } c; c.s = u; return c.b;
}

// ---- CDNA5 async global->LDS copy (16 bytes / lane, ASYNCcnt tracked) -----
// LDS aperture maps generic addr[31:0] directly to the LDS offset, so the low
// 32 bits of a __shared__ pointer are a valid VDST LDS address.
__device__ __forceinline__ void async_copy_b128(unsigned short* lds_dst,
                                                const unsigned short* gsrc) {
  unsigned loff = (unsigned)(unsigned long long)lds_dst;
  asm volatile("global_load_async_to_lds_b128 %0, %1, off"
               :: "v"(loff), "v"(gsrc)
               : "memory");
}
__device__ __forceinline__ void wait_async0() {
  asm volatile("s_wait_asynccnt 0" ::: "memory");
}

// ---- WMMA fragment loaders (CDNA5 ISA §7.12.2 layouts) --------------------
// A-frag 16x32 bf16 from row-major src (ushort elems, leading dim ld).
// lanes 0-15: M=lane, K=0..7 & 16..23 ; lanes 16-31: M=lane-16, K=8..15 & 24..31
__device__ __forceinline__ v16bf load_a_frag(const unsigned short* src, int ld, int lane) {
  const int row = lane & 15;
  const int kh  = (lane >> 4) & 1;
  v16bf a;
#pragma unroll
  for (int i = 0; i < 8; ++i) {
    const int kb = (i < 4) ? (8 * kh + 2 * i) : (16 + 8 * kh + 2 * (i - 4));
    a[2 * i]     = us2bf(src[row * ld + kb]);
    a[2 * i + 1] = us2bf(src[row * ld + kb + 1]);
  }
  return a;
}

// B-frag 32x16 from a row-major [K x N] source (element (k, n) = src[k*ld + n]).
__device__ __forceinline__ v16bf load_b_frag(const unsigned short* src, int ld, int lane) {
  const int col = lane & 15;
  const int kh  = (lane >> 4) & 1;
  v16bf b;
#pragma unroll
  for (int i = 0; i < 8; ++i) {
    const int kb = (i < 4) ? (8 * kh + 2 * i) : (16 + 8 * kh + 2 * (i - 4));
    b[2 * i]     = us2bf(src[kb * ld + col]);
    b[2 * i + 1] = us2bf(src[(kb + 1) * ld + col]);
  }
  return b;
}

// B-frag where the source is stored row-major as [N x K]; computes B = src^T.
// Used for W^T (weights are [out,in]) and for K^T in Q@K^T.
__device__ __forceinline__ v16bf load_bT_frag(const unsigned short* src, int ld, int lane) {
  const int col = lane & 15;
  const int kh  = (lane >> 4) & 1;
  v16bf b;
#pragma unroll
  for (int i = 0; i < 8; ++i) {
    const int kb = (i < 4) ? (8 * kh + 2 * i) : (16 + 8 * kh + 2 * (i - 4));
    b[2 * i]     = us2bf(src[col * ld + kb]);
    b[2 * i + 1] = us2bf(src[col * ld + kb + 1]);
  }
  return b;
}

__device__ __forceinline__ v8f wmma_bf16(v16bf a, v16bf b, v8f c) {
  return __builtin_amdgcn_wmma_f32_16x16x32_bf16(false, a, false, b, (short)0, c, false, false);
}

// ---------------------------------------------------------------------------
// Kernel 0: fp32 -> bf16 convert
// ---------------------------------------------------------------------------
__global__ void cvt_f32_bf16(const float* __restrict__ src,
                             unsigned short* __restrict__ dst, int n) {
  int i = blockIdx.x * blockDim.x + threadIdx.x;
  if (i < n) dst[i] = f2bf_us(src[i]);
}

// ---------------------------------------------------------------------------
// Kernel 1: QKV projection.  y = x @ W^T, written head-major [B,H,S,HD] bf16.
// grid (8 n-tiles == head, 64 m-tiles, 3 = q/k/v), block 256 (8 waves).
// Each block: 128x64 output tile; each wave: 16-row strip x 64 cols (4 accs).
// ---------------------------------------------------------------------------
#define LDX 40
#define LDW 40
__global__ __launch_bounds__(256) void qkv_gemm(
    const unsigned short* __restrict__ xb,
    const unsigned short* __restrict__ wq,
    const unsigned short* __restrict__ wk,
    const unsigned short* __restrict__ wv,
    unsigned short* __restrict__ Qb,
    unsigned short* __restrict__ Kb,
    unsigned short* __restrict__ Vb) {
  __shared__ unsigned short lx[128 * LDX];
  __shared__ unsigned short lw[64 * LDW];

  const int tid  = threadIdx.x;
  const int lane = tid & 31;
  const int wid  = tid >> 5;         // wave 0..7 -> 16-row strip
  const int h    = blockIdx.x;       // head == 64-col tile
  const int m0   = blockIdx.y * 128;

  const unsigned short* W = (blockIdx.z == 0) ? wq : (blockIdx.z == 1) ? wk : wv;
  unsigned short* Out     = (blockIdx.z == 0) ? Qb : (blockIdx.z == 1) ? Kb : Vb;

  v8f zero;
#pragma unroll
  for (int g = 0; g < 8; ++g) zero[g] = 0.0f;
  v8f acc[4];
#pragma unroll
  for (int n = 0; n < 4; ++n) acc[n] = zero;

  const int xrow = tid >> 1, xk = (tid & 1) * 16;   // 128 rows x 32 halfs
  const int wrow = tid >> 2, wk2 = (tid & 3) * 8;   // 64 rows x 32 halfs

  for (int kc = 0; kc < D_ / 32; ++kc) {
    const int k0 = kc * 32;
    // async-stage x tile [128 x 32] and W tile [64 x 32] bf16 into LDS
    async_copy_b128(&lx[xrow * LDX + xk],
                    xb + (size_t)(m0 + xrow) * D_ + k0 + xk);
    async_copy_b128(&lx[xrow * LDX + xk + 8],
                    xb + (size_t)(m0 + xrow) * D_ + k0 + xk + 8);
    async_copy_b128(&lw[wrow * LDW + wk2],
                    W + (size_t)(64 * h + wrow) * D_ + k0 + wk2);
    wait_async0();
    __syncthreads();

    v16bf a = load_a_frag(&lx[(16 * wid) * LDX], LDX, lane);
#pragma unroll
    for (int n = 0; n < 4; ++n) {
      v16bf bf = load_bT_frag(&lw[(16 * n) * LDW], LDW, lane);
      acc[n] = wmma_bf16(a, bf, acc[n]);
    }
    __syncthreads();
  }

  // store into [B,H,S,HD] bf16 (contiguous 64-wide rows -> head == blockIdx.x)
  const int colb = lane & 15;
  const int rowh = lane >> 4;
#pragma unroll
  for (int n = 0; n < 4; ++n) {
#pragma unroll
    for (int g = 0; g < 8; ++g) {
      const int m = m0 + 16 * wid + g + 8 * rowh;
      const int b = m >> 11;          // / S_
      const int s = m & (S_ - 1);
      const int hd = 16 * n + colb;
      Out[(((size_t)(b * H_ + h) * S_) + s) * HD_ + hd] = f2bf_us(acc[n][g]);
    }
  }
}

// ---------------------------------------------------------------------------
// Kernel 2: flash attention.  grid (S/128 q-tiles, H, B), block 256 (8 waves).
// Per block: 128 queries; per wave: 16-query strip, O strip 16x64 (4 f32 accs).
// Streams 64-key K/V tiles through LDS via async copies; scores seeded with
// pos_bias; online softmax with 16-lane reductions matching the WMMA C layout.
// ---------------------------------------------------------------------------
#define LDK 64
#define LDV 64
__global__ __launch_bounds__(256) void attn_flash(
    const unsigned short* __restrict__ Qb,
    const unsigned short* __restrict__ Kb,
    const unsigned short* __restrict__ Vb,
    const float* __restrict__ bias,      // [H,S,S]
    unsigned short* __restrict__ attb) { // [B,S,D] bf16 (merged heads)
  __shared__ unsigned short lk[64 * LDK];     // K tile  [key][hd]
  __shared__ unsigned short lv[64 * LDV];     // V tile  [key][hd]
  __shared__ unsigned short lp[8 * 16 * 64];  // per-wave P scratch [16 q][64 key]

  const int tid  = threadIdx.x;
  const int lane = tid & 31;
  const int wid  = tid >> 5;
  const int q0   = blockIdx.x * 128;
  const int h    = blockIdx.y;
  const int b    = blockIdx.z;
  const int colb = lane & 15;
  const int rowh = lane >> 4;

  const size_t headoff = (size_t)(b * H_ + h) * S_ * HD_;

  // Q strip for this wave lives in registers for the whole kernel.
  const unsigned short* qptr = Qb + headoff + (size_t)(q0 + 16 * wid) * HD_;
  const v16bf qa0 = load_a_frag(qptr, HD_, lane);
  const v16bf qa1 = load_a_frag(qptr + 32, HD_, lane);

  v8f zero;
#pragma unroll
  for (int g = 0; g < 8; ++g) zero[g] = 0.0f;
  v8f oA[4];
#pragma unroll
  for (int n = 0; n < 4; ++n) oA[n] = zero;

  float mrow[8], lrow[8];
#pragma unroll
  for (int g = 0; g < 8; ++g) { mrow[g] = -3.0e38f; lrow[g] = 0.0f; }

  const int trow = tid >> 2, tk = (tid & 3) * 16;  // 64 rows x 64 halfs staging

  for (int kt = 0; kt < S_ / 64; ++kt) {
    const int kbase = kt * 64;
    // async-stage K and V 64x64 bf16 tiles (2x b128 per thread per tile)
    const unsigned short* gk = Kb + headoff + (size_t)(kbase + trow) * HD_ + tk;
    const unsigned short* gv = Vb + headoff + (size_t)(kbase + trow) * HD_ + tk;
    async_copy_b128(&lk[trow * LDK + tk],     gk);
    async_copy_b128(&lk[trow * LDK + tk + 8], gk + 8);
    async_copy_b128(&lv[trow * LDV + tk],     gv);
    async_copy_b128(&lv[trow * LDV + tk + 8], gv + 8);
    // prefetch next k-tile's bias row (gfx1250 global_prefetch_b8 path)
    if (kt + 1 < S_ / 64) {
      const int qp = q0 + 16 * wid + 8 * rowh;
      __builtin_prefetch(&bias[((size_t)h * S_ + qp) * S_ + kbase + 64 + colb], 0, 1);
    }
    wait_async0();
    __syncthreads();

    // ---- S = Q @ K^T + bias ----
    v8f sA[4];
#pragma unroll
    for (int n = 0; n < 4; ++n) {
      v8f c;
#pragma unroll
      for (int g = 0; g < 8; ++g) {
        const int q  = q0 + 16 * wid + g + 8 * rowh;
        const int kk = kbase + 16 * n + colb;
        c[g] = bias[((size_t)h * S_ + q) * S_ + kk];
      }
      v16bf bk0 = load_bT_frag(&lk[(16 * n) * LDK + 0],  LDK, lane);
      c = wmma_bf16(qa0, bk0, c);
      v16bf bk1 = load_bT_frag(&lk[(16 * n) * LDK + 32], LDK, lane);
      c = wmma_bf16(qa1, bk1, c);
      sA[n] = c;
    }

    // ---- online softmax (row spans 16 lanes of one VGPR) ----
#pragma unroll
    for (int g = 0; g < 8; ++g) {
      float t = fmaxf(fmaxf(sA[0][g], sA[1][g]), fmaxf(sA[2][g], sA[3][g]));
#pragma unroll
      for (int msk = 1; msk < 16; msk <<= 1) t = fmaxf(t, __shfl_xor(t, msk, 32));
      const float mnew  = fmaxf(mrow[g], t);
      const float scale = __expf(mrow[g] - mnew);
      float rs = 0.0f;
#pragma unroll
      for (int n = 0; n < 4; ++n) { sA[n][g] = __expf(sA[n][g] - mnew); rs += sA[n][g]; }
#pragma unroll
      for (int msk = 1; msk < 16; msk <<= 1) rs += __shfl_xor(rs, msk, 32);
      lrow[g] = lrow[g] * scale + rs;
#pragma unroll
      for (int n = 0; n < 4; ++n) oA[n][g] *= scale;
      mrow[g] = mnew;
    }

    // ---- P -> bf16 A-layout via per-wave LDS scratch ----
    unsigned short* myp = &lp[wid * (16 * 64)];
#pragma unroll
    for (int n = 0; n < 4; ++n) {
#pragma unroll
      for (int g = 0; g < 8; ++g) {
        myp[(g + 8 * rowh) * 64 + 16 * n + colb] = f2bf_us(sA[n][g]);
      }
    }
    const v16bf pa0 = load_a_frag(myp, 64, lane);
    const v16bf pa1 = load_a_frag(myp + 32, 64, lane);

    // ---- O += P @ V ----
#pragma unroll
    for (int n = 0; n < 4; ++n) {
      v16bf vb0 = load_b_frag(&lv[0 * LDV + 16 * n],  LDV, lane);
      oA[n] = wmma_bf16(pa0, vb0, oA[n]);
      v16bf vb1 = load_b_frag(&lv[32 * LDV + 16 * n], LDV, lane);
      oA[n] = wmma_bf16(pa1, vb1, oA[n]);
    }
    __syncthreads();
  }

  // ---- normalize and store merged [B,S,D] bf16 ----
  float inv[8];
#pragma unroll
  for (int g = 0; g < 8; ++g) inv[g] = 1.0f / lrow[g];
#pragma unroll
  for (int n = 0; n < 4; ++n) {
#pragma unroll
    for (int g = 0; g < 8; ++g) {
      const int s = q0 + 16 * wid + g + 8 * rowh;
      const int d = h * HD_ + 16 * n + colb;
      attb[((size_t)b * S_ + s) * D_ + d] = f2bf_us(oA[n][g] * inv[g]);
    }
  }
}

// ---------------------------------------------------------------------------
// Kernel 3: output projection.  out = att @ Wo^T, fp32 result.
// grid (8 n-tiles, 64 m-tiles), block 256.
// ---------------------------------------------------------------------------
__global__ __launch_bounds__(256) void out_gemm(
    const unsigned short* __restrict__ attb,
    const unsigned short* __restrict__ wob,
    float* __restrict__ out) {
  __shared__ unsigned short lx[128 * LDX];
  __shared__ unsigned short lw[64 * LDW];

  const int tid  = threadIdx.x;
  const int lane = tid & 31;
  const int wid  = tid >> 5;
  const int n0   = blockIdx.x * 64;
  const int m0   = blockIdx.y * 128;

  v8f zero;
#pragma unroll
  for (int g = 0; g < 8; ++g) zero[g] = 0.0f;
  v8f acc[4];
#pragma unroll
  for (int n = 0; n < 4; ++n) acc[n] = zero;

  const int xrow = tid >> 1, xk = (tid & 1) * 16;
  const int wrow = tid >> 2, wk2 = (tid & 3) * 8;

  for (int kc = 0; kc < D_ / 32; ++kc) {
    const int k0 = kc * 32;
    async_copy_b128(&lx[xrow * LDX + xk],
                    attb + (size_t)(m0 + xrow) * D_ + k0 + xk);
    async_copy_b128(&lx[xrow * LDX + xk + 8],
                    attb + (size_t)(m0 + xrow) * D_ + k0 + xk + 8);
    async_copy_b128(&lw[wrow * LDW + wk2],
                    wob + (size_t)(n0 + wrow) * D_ + k0 + wk2);
    wait_async0();
    __syncthreads();

    v16bf a = load_a_frag(&lx[(16 * wid) * LDX], LDX, lane);
#pragma unroll
    for (int n = 0; n < 4; ++n) {
      v16bf bf = load_bT_frag(&lw[(16 * n) * LDW], LDW, lane);
      acc[n] = wmma_bf16(a, bf, acc[n]);
    }
    __syncthreads();
  }

  const int colb = lane & 15;
  const int rowh = lane >> 4;
#pragma unroll
  for (int n = 0; n < 4; ++n) {
#pragma unroll
    for (int g = 0; g < 8; ++g) {
      const int m = m0 + 16 * wid + g + 8 * rowh;
      out[(size_t)m * D_ + n0 + 16 * n + colb] = acc[n][g];
    }
  }
}

// ---------------------------------------------------------------------------
extern "C" void kernel_launch(void* const* d_in, const int* in_sizes, int n_in,
                              void* d_out, int out_size, void* d_ws, size_t ws_size,
                              hipStream_t stream) {
  (void)in_sizes; (void)n_in; (void)out_size; (void)ws_size;

  const float* x  = (const float*)d_in[0];
  const float* pb = (const float*)d_in[1];
  const float* Wq = (const float*)d_in[2];
  const float* Wk = (const float*)d_in[3];
  const float* Wv = (const float*)d_in[4];
  const float* Wo = (const float*)d_in[5];
  float* out = (float*)d_out;

  // Workspace carve-up (~42 MB total, all bf16 stored as ushort)
  char* ws = (char*)d_ws;
  size_t off = 0;
  auto carve = [&](size_t nhalf) {
    unsigned short* p = (unsigned short*)(ws + off);
    off += nhalf * sizeof(unsigned short);
    return p;
  };
  unsigned short* xb   = carve((size_t)M_ * D_);
  unsigned short* wqb  = carve((size_t)D_ * D_);
  unsigned short* wkb  = carve((size_t)D_ * D_);
  unsigned short* wvb  = carve((size_t)D_ * D_);
  unsigned short* wob  = carve((size_t)D_ * D_);
  unsigned short* Qb   = carve((size_t)B_ * H_ * S_ * HD_);
  unsigned short* Kb   = carve((size_t)B_ * H_ * S_ * HD_);
  unsigned short* Vb   = carve((size_t)B_ * H_ * S_ * HD_);
  unsigned short* attb = carve((size_t)M_ * D_);

  const int nx = M_ * D_;        // 4,194,304
  const int nw = D_ * D_;        // 262,144
  cvt_f32_bf16<<<(nx + 255) / 256, 256, 0, stream>>>(x,  xb,  nx);
  cvt_f32_bf16<<<(nw + 255) / 256, 256, 0, stream>>>(Wq, wqb, nw);
  cvt_f32_bf16<<<(nw + 255) / 256, 256, 0, stream>>>(Wk, wkb, nw);
  cvt_f32_bf16<<<(nw + 255) / 256, 256, 0, stream>>>(Wv, wvb, nw);
  cvt_f32_bf16<<<(nw + 255) / 256, 256, 0, stream>>>(Wo, wob, nw);

  qkv_gemm<<<dim3(H_, M_ / 128, 3), 256, 0, stream>>>(xb, wqb, wkb, wvb, Qb, Kb, Vb);

  attn_flash<<<dim3(S_ / 128, H_, B_), 256, 0, stream>>>(Qb, Kb, Vb, pb, attb);

  out_gemm<<<dim3(D_ / 64, M_ / 128), 256, 0, stream>>>(attb, wob, out);
}
